// MHCrossAttn_81750407512809
// MI455X (gfx1250) — matrix operand order repaired and checked by reading, not compiled, not run on hardware
//
#include <hip/hip_runtime.h>
#include <hip/hip_bf16.h>
#include <stdint.h>

// ---------------- problem constants ----------------
#define BB   2
#define SS   2048
#define DD   512
#define HH   8
#define DKK  64
#define KP1  16
#define RR   32
#define BS   (BB * SS)           // 4096 rows in the projection GEMMs

typedef __attribute__((ext_vector_type(16))) _Float16 v16h;
typedef __attribute__((ext_vector_type(8)))  _Float16 v8h;
typedef __attribute__((ext_vector_type(8)))  float    v8f;
typedef __attribute__((ext_vector_type(4)))  unsigned v4u;
typedef __attribute__((ext_vector_type(8)))  unsigned v8u;

union HFrag { v16h v; unsigned u[8]; };

// ---------------- utility kernels ----------------
__global__ void f32_to_f16_k(const float* __restrict__ src,
                             _Float16* __restrict__ dst, int n) {
  int i = blockIdx.x * blockDim.x + threadIdx.x;
  if (i < n) dst[i] = (_Float16)src[i];
}

__global__ void zero_f32_k(float* __restrict__ p, int n) {
  int i = blockIdx.x * blockDim.x + threadIdx.x;
  if (i < n) p[i] = 0.0f;
}

// ---------------- Tensor Data Mover: 64x32-half 2D tile -> LDS ----------------
// Builds the 4-group D# (CDNA5 ISA) and issues
// tensor_load_to_lds. data_size=2B, tile = 32 elems x 64 rows cut from a
// row-major [M,K] f16 tensor. Issued by one wave per block; tracked by
// TENSORcnt.
__device__ __forceinline__ void tdm_load_Atile(const _Float16* A, int M, int K,
                                               int mblock, int k0,
                                               unsigned ldsByteOff) {
  unsigned long long ga =
      (unsigned long long)(uintptr_t)(A + (size_t)mblock * K + k0);
  v4u g0;
  g0.x = 1u;                                        // count=1, user descriptor
  g0.y = ldsByteOff;                                // LDS byte address
  g0.z = (unsigned)(ga & 0xFFFFFFFFu);              // global_addr[31:0]
  g0.w = (unsigned)((ga >> 32) & 0x01FFFFFFu) | (2u << 30);  // addr[56:32],type=2
  v8u g1;
  g1.s0 = 0x00010000u;                              // data_size=1 (2B), no mcast
  g1.s1 = ((unsigned)K & 0xFFFFu) << 16;            // tensor_dim0[15:0] @bit48
  g1.s2 = (((unsigned)K >> 16) & 0xFFFFu)           // tensor_dim0[31:16]
        | (((unsigned)M & 0xFFFFu) << 16);          // tensor_dim1[15:0] @bit80
  g1.s3 = (((unsigned)M >> 16) & 0xFFFFu)           // tensor_dim1[31:16]
        | (32u << 16);                              // tile_dim0 = 32 @bit112
  g1.s4 = 64u;                                      // tile_dim1=64, tile_dim2=0
  g1.s5 = (unsigned)K;                              // tensor_dim0_stride[31:0]
  g1.s6 = 0u;                                       // stride0 hi, stride1 lo
  g1.s7 = 0u;
  v4u g2 = {0u, 0u, 0u, 0u};                        // 2D: dims 2/3 unused
  v4u g3 = {0u, 0u, 0u, 0u};
  asm volatile("tensor_load_to_lds %0, %1, %2, %3"
               :: "s"(g0), "s"(g1), "s"(g2), "s"(g3)
               : "memory");
}

// ---------------- WMMA GEMM ----------------
// C[M,N] = A[M,K] * W[N,K]^T + bias[N]   (torch Linear convention)
// Block: 64 M-rows x 128 N-cols, 8 waves; each wave owns a 64x16 strip
// (4 accumulators), reusing one B fragment across 4 WMMAs per K-step.
// A tile (64x32 f16 = 4KB) is double-buffered in LDS, filled by the TDM.
// EPI==0: store f16, split-head layout [B,H,S,DK]   (QKV projections)
// EPI==1: store f32, row-major [M,N]                (output projection)
template <int EPI>
__global__ __launch_bounds__(256)
void gemm_wmma_k(const _Float16* __restrict__ A,
                 const _Float16* __restrict__ W,
                 const float*    __restrict__ bias,
                 void*           __restrict__ dst,
                 int M, int N, int K) {
  __shared__ _Float16 ldsA[2][64 * 32];    // first (only) LDS alloc -> offset 0
  const unsigned tileBytes = 64u * 32u * 2u;

  const int lane   = threadIdx.x & 31;
  const int waveId = threadIdx.x >> 5;
  const int hf     = lane >> 4;            // lane-half select
  const int l15    = lane & 15;
  const int mblock = blockIdx.x * 64;
  const int col    = blockIdx.y * 128 + waveId * 16 + l15;

  const int nsteps = K / 32;
  v8f c[4] = {{}, {}, {}, {}};

  if (waveId == 0) {                       // prime buffer 0
    tdm_load_Atile(A, M, K, mblock, 0, 0u);
  }

  for (int i = 0; i < nsteps; ++i) {
    const int k0 = i * 32;
    const int buf = i & 1;
    if (waveId == 0) {
      if (i + 1 < nsteps) {
        tdm_load_Atile(A, M, K, mblock, k0 + 32,
                       ((i + 1) & 1) ? tileBytes : 0u);
        __builtin_amdgcn_s_wait_tensorcnt(1);   // buf[i&1] resident
      } else {
        __builtin_amdgcn_s_wait_tensorcnt(0);
      }
    }
    __syncthreads();                       // tile visible to all waves

    // B fragment: VGPR j holds K pair (hf*16 + 2j) for column `col`
    HFrag b;
#pragma unroll
    for (int j = 0; j < 8; ++j) {
      b.u[j] = *(const unsigned*)(W + (size_t)col * K + k0 + hf * 16 + 2 * j);
    }
    if (k0 + 32 < K) {
      __builtin_prefetch(W + (size_t)col * K + k0 + 32, 0, 3);
    }

    const unsigned* lp = (const unsigned*)&ldsA[buf][0];
#pragma unroll
    for (int t = 0; t < 4; ++t) {
      // A frag for rows [16t,16t+16): VGPR j holds K pair
      // (hf*16 + 2j) of row (16t + l15)
      HFrag a;
#pragma unroll
      for (int j = 0; j < 8; ++j) {
        const int ku = hf * 8 + j;
        a.u[j] = lp[(16 * t + l15) * 16 + ku];
      }
      c[t] = __builtin_amdgcn_wmma_f32_16x16x32_f16(
          false, a.v, false, b.v, (short)0, c[t], false, false);
    }
    __syncthreads();                       // all reads done before overwrite
  }

  const float bval = bias[col];
#pragma unroll
  for (int t = 0; t < 4; ++t) {
#pragma unroll
    for (int i = 0; i < 8; ++i) {
      const int m   = mblock + 16 * t + i + 8 * hf;
      const float v = c[t][i] + bval;
      if (EPI == 0) {
        const int bi = m >> 11, s = m & (SS - 1);
        const int h  = col >> 6, dk = col & (DKK - 1);
        ((_Float16*)dst)[(((size_t)bi * HH + h) * SS + s) * DKK + dk] =
            (_Float16)v;
      } else {
        ((float*)dst)[(size_t)m * N + col] = v;
      }
    }
  }
}

// ---------------- attention core ----------------
__device__ __forceinline__ void decode_idx(int idx, int& bb, int& h, int& r,
                                           int& s) {
  s = idx & (SS - 1);
  int t = idx >> 11;
  r = t & (RR - 1);
  t >>= 5;
  h = t & (HH - 1);
  bb = t >> 3;
}

__device__ __forceinline__ void gather_indices(
    int bb, int h, int r, int s,
    const int* __restrict__ pos_enc,  const int* __restrict__ pos_pad,
    const int* __restrict__ cpos_enc, const int* __restrict__ cpos_pad,
    int& qidx, int& kvidx, bool& mask) {
  const int rr = r & 15;
  const size_t off = (((size_t)bb * HH + h) * KP1 + rr) * SS + s;
  const int qpe_raw = cpos_enc[off];
  const bool qmask  = (qpe_raw == -1);
  const int qpe     = qmask ? 0 : qpe_raw;
  qidx = (r < KP1) ? cpos_pad[off] : qpe;          // q_idx = [q_pos_pad ; q_pe]
  const int kpe_raw = pos_enc[off];
  const int kpe     = (kpe_raw == -1) ? 0 : kpe_raw;
  kvidx = (r < KP1) ? kpe : pos_pad[off];          // kv_idx = [kv_pe ; kv_pos_pad]
  mask = qmask || (r == 0);                        // first S flat entries masked
}

__global__ __launch_bounds__(256)
void scores_k(const _Float16* __restrict__ qh, const _Float16* __restrict__ kh,
              const int* __restrict__ pos_enc,  const int* __restrict__ pos_pad,
              const int* __restrict__ cpos_enc, const int* __restrict__ cpos_pad,
              const float* __restrict__ rel_q,  const float* __restrict__ rel_k,
              float* __restrict__ scoresExp, float* __restrict__ denom) {
  const int idx = blockIdx.x * blockDim.x + threadIdx.x;   // over B*H*R*S
  int bb, h, r, s;  decode_idx(idx, bb, h, r, s);
  int qidx, kvidx;  bool mask;
  gather_indices(bb, h, r, s, pos_enc, pos_pad, cpos_enc, cpos_pad,
                 qidx, kvidx, mask);

  const v8h* q8 = (const v8h*)(qh + (((size_t)bb * HH + h) * SS + qidx)  * DKK);
  const v8h* k8 = (const v8h*)(kh + (((size_t)bb * HH + h) * SS + kvidx) * DKK);
  const float* rq = rel_q + ((size_t)h * RR + r) * DKK;
  const float* rk = rel_k + ((size_t)h * RR + r) * DKK;

  float acc = 0.0f;
#pragma unroll
  for (int cTile = 0; cTile < 8; ++cTile) {        // 8 x 16B vector loads / row
    const v8h qv = q8[cTile];
    const v8h kv = k8[cTile];
#pragma unroll
    for (int e = 0; e < 8; ++e) {
      const float qf = (float)qv[e];
      const float kf = (float)kv[e];
      const int d = cTile * 8 + e;
      acc += qf * kf + qf * rk[d] + rq[d] * kf;    // c2c + c2p + p2c
    }
  }
  acc *= (1.0f / 24.0f);                           // 1 / (3*sqrt(64))
  const float w = mask ? 0.0f : __expf(acc);
  scoresExp[idx] = w;
  if (w != 0.0f) {
    atomicAdd(&denom[((size_t)bb * HH + h) * SS + qidx], w);
  }
}

// one thread per (element, d-channel): 64 consecutive lanes cover one gathered
// row -> value loads and the f32 atomic scatter both coalesce per cacheline.
__global__ __launch_bounds__(256)
void scatter_k(const _Float16* __restrict__ vh,
               const int* __restrict__ pos_enc,  const int* __restrict__ pos_pad,
               const int* __restrict__ cpos_enc, const int* __restrict__ cpos_pad,
               const float* __restrict__ rel_v,
               const float* __restrict__ scoresExp, const float* __restrict__ denom,
               float* __restrict__ attnOut) {
  const long long tid =
      (long long)blockIdx.x * blockDim.x + threadIdx.x;    // over B*H*R*S*DK
  const int d   = (int)(tid & (DKK - 1));
  const int idx = (int)(tid >> 6);
  int bb, h, r, s;  decode_idx(idx, bb, h, r, s);
  int qidx, kvidx;  bool mask;
  gather_indices(bb, h, r, s, pos_enc, pos_pad, cpos_enc, cpos_pad,
                 qidx, kvidx, mask);

  const float wgt = scoresExp[idx];
  float den = denom[((size_t)bb * HH + h) * SS + qidx];
  if (den == 0.0f) den = -1e9f;                    // reference empty-bucket fixup
  const float wn = wgt / den;
  if (wn == 0.0f) return;

  const float vv = (float)vh[(((size_t)bb * HH + h) * SS + kvidx) * DKK + d];
  const float rv = rel_v[((size_t)h * RR + r) * DKK + d];
  atomicAdd(&attnOut[(((size_t)bb * HH + h) * SS + qidx) * DKK + d],
            wn * (vv + rv));
}

__global__ void merge_heads_k(const float* __restrict__ attnOut,
                              _Float16* __restrict__ merged) {
  const int i = blockIdx.x * blockDim.x + threadIdx.x;     // over B*S*D
  const int e = i & (DD - 1);
  int t = i >> 9;
  const int s  = t & (SS - 1);
  const int bb = t >> 11;
  const int h  = e >> 6, dk = e & (DKK - 1);
  merged[i] = (_Float16)attnOut[(((size_t)bb * HH + h) * SS + s) * DKK + dk];
}

// ---------------- launch ----------------
extern "C" void kernel_launch(void* const* d_in, const int* in_sizes, int n_in,
                              void* d_out, int out_size, void* d_ws, size_t ws_size,
                              hipStream_t stream) {
  const float* query = (const float*)d_in[0];
  const float* key   = (const float*)d_in[1];
  const float* value = (const float*)d_in[2];
  const int* pos_enc  = (const int*)d_in[3];
  const int* pos_pad  = (const int*)d_in[4];
  const int* cpos_enc = (const int*)d_in[5];
  const int* cpos_pad = (const int*)d_in[6];
  const float* rel_q = (const float*)d_in[7];
  const float* rel_k = (const float*)d_in[8];
  const float* rel_v = (const float*)d_in[9];
  const float* Wq = (const float*)d_in[10]; const float* bq = (const float*)d_in[11];
  const float* Wk = (const float*)d_in[12]; const float* bk = (const float*)d_in[13];
  const float* Wv = (const float*)d_in[14]; const float* bv = (const float*)d_in[15];
  const float* Wo = (const float*)d_in[16]; const float* bo = (const float*)d_in[17];

  // scratch carve-up (256B aligned)
  char* w = (char*)d_ws;
  auto carve = [&](size_t bytes) -> void* {
    void* p = (void*)w;
    w += (bytes + 255) & ~(size_t)255;
    return p;
  };
  const size_t nInp = (size_t)BS * DD;            // 2,097,152
  const size_t nW   = (size_t)DD * DD;            // 262,144
  const size_t nRS  = (size_t)BB * HH * RR * SS;  // 1,048,576
  _Float16* hq  = (_Float16*)carve(nInp * 2);
  _Float16* hk  = (_Float16*)carve(nInp * 2);
  _Float16* hv  = (_Float16*)carve(nInp * 2);
  _Float16* hWq = (_Float16*)carve(nW * 2);
  _Float16* hWk = (_Float16*)carve(nW * 2);
  _Float16* hWv = (_Float16*)carve(nW * 2);
  _Float16* hWo = (_Float16*)carve(nW * 2);
  _Float16* qh  = (_Float16*)carve(nInp * 2);     // [B,H,S,DK] f16
  _Float16* kh  = (_Float16*)carve(nInp * 2);
  _Float16* vh  = (_Float16*)carve(nInp * 2);
  float* scoresExp = (float*)carve(nRS * 4);
  float* denom     = (float*)carve((size_t)BB * HH * SS * 4);
  float* attnOut   = (float*)carve(nInp * 4);     // [B,H,S,DK] f32 accum
  _Float16* merged = (_Float16*)carve(nInp * 2);  // [B,S,D] f16

  const int T = 256;
  // 1) down-convert inputs + weights to f16
  f32_to_f16_k<<<(int)(nInp / T), T, 0, stream>>>(query, hq, (int)nInp);
  f32_to_f16_k<<<(int)(nInp / T), T, 0, stream>>>(key,   hk, (int)nInp);
  f32_to_f16_k<<<(int)(nInp / T), T, 0, stream>>>(value, hv, (int)nInp);
  f32_to_f16_k<<<(int)(nW / T),   T, 0, stream>>>(Wq, hWq, (int)nW);
  f32_to_f16_k<<<(int)(nW / T),   T, 0, stream>>>(Wk, hWk, (int)nW);
  f32_to_f16_k<<<(int)(nW / T),   T, 0, stream>>>(Wv, hWv, (int)nW);
  f32_to_f16_k<<<(int)(nW / T),   T, 0, stream>>>(Wo, hWo, (int)nW);

  // 2) QKV projection GEMMs (WMMA + TDM double-buffered A tiles)
  dim3 gemmGrid(BS / 64, DD / 128);
  gemm_wmma_k<0><<<gemmGrid, T, 0, stream>>>(hq, hWq, bq, qh, BS, DD, DD);
  gemm_wmma_k<0><<<gemmGrid, T, 0, stream>>>(hk, hWk, bk, kh, BS, DD, DD);
  gemm_wmma_k<0><<<gemmGrid, T, 0, stream>>>(hv, hWv, bv, vh, BS, DD, DD);

  // 3) zero accumulators (harness does not re-zero between replays)
  const int nDen = BB * HH * SS;
  zero_f32_k<<<(nDen + T - 1) / T, T, 0, stream>>>(denom, nDen);
  zero_f32_k<<<(int)(nInp / T), T, 0, stream>>>(attnOut, (int)nInp);

  // 4) scores: gather + 3 dots + masked exp + scatter-add denominator
  scores_k<<<(int)(nRS / T), T, 0, stream>>>(qh, kh, pos_enc, pos_pad,
                                             cpos_enc, cpos_pad, rel_q, rel_k,
                                             scoresExp, denom);
  // 5) normalize + coalesced weighted scatter-add of (v + rel_v)
  scatter_k<<<(int)(nRS * DKK / T), T, 0, stream>>>(vh, pos_enc, pos_pad,
                                                    cpos_enc, cpos_pad, rel_v,
                                                    scoresExp, denom, attnOut);
  // 6) merge heads -> f16 [B,S,D]
  merge_heads_k<<<(int)(nInp / T), T, 0, stream>>>(attnOut, merged);

  // 7) output projection (WMMA), f32 straight into d_out
  gemm_wmma_k<1><<<gemmGrid, T, 0, stream>>>(merged, hWo, bo, (float*)d_out,
                                             BS, DD, DD);
}
